// SmearAdapterLayer_53008486367834
// MI455X (gfx1250) — matrix operand order repaired and checked by reading, not compiled, not run on hardware
//
#include <hip/hip_runtime.h>
#include <hip/hip_bf16.h>
#include <math.h>

#define Bdim 4
#define Sdim 2048
#define BS   (Bdim * Sdim)     // 8192 tokens
#define Hdim 768
#define H2   384
#define Ddim 1024
#define Edim 8

typedef __bf16 bf16;
typedef __attribute__((ext_vector_type(8)))  __bf16 v8bf;
typedef __attribute__((ext_vector_type(16))) __bf16 v16bf;
typedef __attribute__((ext_vector_type(8)))  float  v8f;

__device__ __forceinline__ float gelu_exact(float x) {
    return 0.5f * x * (1.0f + erff(x * 0.70710678118654752440f));
}

// ---- WMMA fragment loaders per CDNA5 ISA 7.12.2 (wave32) ----
// A: 16x32 bf16, row-major source with leading dim ldk (elements).
// lane l: row = l&15 ; K halves: [kb, kb+8) and [kb+16, kb+24), kb = k0 + (l&16 ? 8 : 0)
__device__ __forceinline__ v16bf load_frag_a(const bf16* base, int ldk, int k0, int lane) {
    int m  = lane & 15;
    int kb = k0 + ((lane & 16) ? 8 : 0);
    const bf16* p = base + (size_t)m * ldk + kb;
    v8bf lo = *(const v8bf*)(p);
    v8bf hi = *(const v8bf*)(p + 16);
    return __builtin_shufflevector(lo, hi, 0,1,2,3,4,5,6,7,8,9,10,11,12,13,14,15);
}
// B: 32x16 bf16 from an N-major weight (layout [N][K], leading dim ldk).
// lane l: col = n0 + (l&15) ; K halves: [kb, kb+8) and [kb+16, kb+24), kb = k0 + (l&16 ? 8 : 0)
__device__ __forceinline__ v16bf load_frag_b(const bf16* base, int ldk, int n0, int k0, int lane) {
    int n  = n0 + (lane & 15);
    int kb = k0 + ((lane & 16) ? 8 : 0);
    const bf16* p = base + (size_t)n * ldk + kb;
    v8bf lo = *(const v8bf*)(p);
    v8bf hi = *(const v8bf*)(p + 16);
    return __builtin_shufflevector(lo, hi, 0,1,2,3,4,5,6,7,8,9,10,11,12,13,14,15);
}

// ---------------- conversion kernels ----------------
__global__ __launch_bounds__(256) void k_cvt_bf16(const float* __restrict__ src,
                                                  bf16* __restrict__ dst, int n) {
    int i = blockIdx.x * 256 + threadIdx.x;
    if (i < n) dst[i] = (bf16)src[i];
}

// src [R][C] fp32 -> dst [C][R] bf16   (single matrix)
__global__ __launch_bounds__(256) void k_cvt_transpose(const float* __restrict__ src,
                                                       bf16* __restrict__ dst, int R, int C) {
    int i = blockIdx.x * 256 + threadIdx.x;
    if (i < R * C) {
        int r = i / C, c = i % C;
        dst[(size_t)c * R + r] = (bf16)src[i];
    }
}

// batched: src [E][R][C] -> dst [E][C][R], blockIdx.y = expert
__global__ __launch_bounds__(256) void k_cvt_transpose_b(const float* __restrict__ src,
                                                         bf16* __restrict__ dst, int R, int C) {
    int i = blockIdx.x * 256 + threadIdx.x;
    size_t o = (size_t)blockIdx.y * R * C;
    if (i < R * C) {
        int r = i / C, c = i % C;
        dst[o + (size_t)c * R + r] = (bf16)src[o + i];
    }
}

// ---------------- router GEMM: h1 = gelu(x @ w1 + b1), M=8192 N=384 K=768 ----------------
__global__ __launch_bounds__(256) void k_router_gemm(const bf16* __restrict__ xb,
                                                     const bf16* __restrict__ w1T, // [384][768]
                                                     const float* __restrict__ b1,
                                                     float* __restrict__ h1) {
    __shared__ bf16 Xs[16 * Hdim];   // 24 KB
    const int tid = threadIdx.x;
    const int m0  = blockIdx.x * 16;
    // tile rows are contiguous in xb: straight copy of 16*768*2 bytes
    {
        const uint32_t* s = (const uint32_t*)(xb + (size_t)m0 * Hdim);
        uint32_t* d = (uint32_t*)Xs;
        for (int i = tid; i < 16 * Hdim / 2; i += 256) d[i] = s[i];
    }
    __syncthreads();
    const int wave = tid >> 5, lane = tid & 31;
    const int n0 = (blockIdx.y * 8 + wave) * 16;
    v8f acc = {};
    for (int kt = 0; kt < Hdim / 32; ++kt) {
        v16bf a = load_frag_a(Xs, Hdim, kt * 32, lane);
        v16bf b = load_frag_b(w1T, Hdim, n0, kt * 32, lane);
        acc = __builtin_amdgcn_wmma_f32_16x16x32_bf16(false, a, false, b, (short)0, acc, false, false);
    }
    const int col  = n0 + (lane & 15);
    const int rsel = (lane & 16) ? 8 : 0;
    const float bias = b1[col];
    for (int v = 0; v < 8; ++v) {
        int row = m0 + v + rsel;
        h1[(size_t)row * H2 + col] = gelu_exact(acc[v] + bias);
    }
}

// ---------------- router: logits, softmax, top-2, compaction ----------------
__global__ __launch_bounds__(256) void k_router_topk(const float* __restrict__ h1,
                                                     const float* __restrict__ w2, // [384][8]
                                                     const float* __restrict__ b2,
                                                     int* __restrict__ counts,
                                                     int* __restrict__ tok_list,
                                                     float* __restrict__ gate_list) {
    int t = blockIdx.x * 256 + threadIdx.x;
    if (t >= BS) return;
    float acc[Edim];
    #pragma unroll
    for (int e = 0; e < Edim; ++e) acc[e] = b2[e];
    const float* hr = h1 + (size_t)t * H2;
    for (int k = 0; k < H2; ++k) {
        float hk = hr[k];
        const float* wr = w2 + k * Edim;
        #pragma unroll
        for (int e = 0; e < Edim; ++e) acc[e] += hk * wr[e];
    }
    float mx = acc[0];
    #pragma unroll
    for (int e = 1; e < Edim; ++e) mx = fmaxf(mx, acc[e]);
    float s = 0.f;
    #pragma unroll
    for (int e = 0; e < Edim; ++e) { acc[e] = __expf(acc[e] - mx); s += acc[e]; }
    // top-2 (ties -> lowest index, matching lax.top_k)
    int i0 = 0;
    #pragma unroll
    for (int e = 1; e < Edim; ++e) if (acc[e] > acc[i0]) i0 = e;
    int i1 = (i0 == 0) ? 1 : 0;
    #pragma unroll
    for (int e = 0; e < Edim; ++e) if (e != i0 && acc[e] > acc[i1]) i1 = e;
    float p0 = acc[i0] / s, p1 = acc[i1] / s;
    float inv = 1.0f / (p0 + p1 + 1e-8f);
    int s0 = atomicAdd(&counts[i0], 1);
    tok_list[i0 * BS + s0] = t; gate_list[i0 * BS + s0] = p0 * inv;
    int s1 = atomicAdd(&counts[i1], 1);
    tok_list[i1 * BS + s1] = t; gate_list[i1 * BS + s1] = p1 * inv;
}

// ---------------- expert FFN over compacted token lists ----------------
__global__ __launch_bounds__(256) void k_expert_ffn(const bf16* __restrict__ xb,
                                                    const bf16* __restrict__ dwT, // [E][D][H]
                                                    const bf16* __restrict__ upT, // [E][H][D]
                                                    const float* __restrict__ down_b, // [E][D]
                                                    const int* __restrict__ counts,
                                                    const int* __restrict__ tok_list,
                                                    const float* __restrict__ gate_list,
                                                    float* __restrict__ accbuf) {
    __shared__ bf16 Xs[16 * Hdim];   // 24 KB
    __shared__ bf16 Zs[16 * Ddim];   // 32 KB
    __shared__ int   tok[16];
    __shared__ float gv[16];
    const int e = blockIdx.y;
    const int cnt = counts[e];
    const int start = blockIdx.x * 16;
    if (start >= cnt) return;                       // block-uniform exit
    const int tid = threadIdx.x;
    if (tid < 16) {
        int slot = start + tid;
        tok[tid] = (slot < cnt) ? tok_list[e * BS + slot] : -1;
        gv[tid]  = (slot < cnt) ? gate_list[e * BS + slot] : 0.0f;
    }
    __syncthreads();
    // gather 16 token rows (bf16) into LDS
    for (int i = tid; i < 16 * Hdim / 2; i += 256) {
        int r = i / (Hdim / 2), c = i % (Hdim / 2);
        uint32_t v = 0;
        int tk = tok[r];
        if (tk >= 0) v = ((const uint32_t*)(xb + (size_t)tk * Hdim))[c];
        ((uint32_t*)Xs)[i] = v;
    }
    __syncthreads();
    const int wave = tid >> 5, lane = tid & 31;
    const bf16* Wd = dwT + (size_t)e * Hdim * Ddim;  // [D][H]  N-major, K=768
    const bf16* Wu = upT + (size_t)e * Hdim * Ddim;  // [H][D]  N-major, K=1024
    const int col_lo = lane & 15;
    const int rsel   = (lane & 16) ? 8 : 0;
    // phase 1: Z = gelu(X @ Wdn + bdn)  -> 16x1024, 64 n-tiles over 8 waves
    for (int j = 0; j < 8; ++j) {
        int n0 = (wave + 8 * j) * 16;
        v8f acc = {};
        for (int kt = 0; kt < Hdim / 32; ++kt) {
            v16bf a = load_frag_a(Xs, Hdim, kt * 32, lane);
            v16bf b = load_frag_b(Wd, Hdim, n0, kt * 32, lane);
            acc = __builtin_amdgcn_wmma_f32_16x16x32_bf16(false, a, false, b, (short)0, acc, false, false);
        }
        int col = n0 + col_lo;
        float bias = down_b[e * Ddim + col];
        #pragma unroll
        for (int v = 0; v < 8; ++v)
            Zs[(size_t)(v + rsel) * Ddim + col] = (bf16)gelu_exact(acc[v] + bias);
    }
    __syncthreads();
    // phase 2: Y = Z @ Wup -> 16x768, 48 n-tiles over 8 waves; gate & scatter-add
    for (int j = 0; j < 6; ++j) {
        int n0 = (wave + 8 * j) * 16;
        v8f acc = {};
        for (int kt = 0; kt < Ddim / 32; ++kt) {
            v16bf a = load_frag_a(Zs, Ddim, kt * 32, lane);
            v16bf b = load_frag_b(Wu, Ddim, n0, kt * 32, lane);
            acc = __builtin_amdgcn_wmma_f32_16x16x32_bf16(false, a, false, b, (short)0, acc, false, false);
        }
        int col = n0 + col_lo;
        #pragma unroll
        for (int v = 0; v < 8; ++v) {
            int row = v + rsel;
            int tk = tok[row];
            if (tk >= 0)
                atomicAdd(&accbuf[(size_t)tk * Hdim + col], gv[row] * acc[v]);
        }
    }
}

// ---------------- residual + LayerNorm, one block per token ----------------
__global__ __launch_bounds__(256) void k_residual_ln(const float* __restrict__ x,
                                                     const float* __restrict__ accbuf,
                                                     const float* __restrict__ gamma,
                                                     const float* __restrict__ beta,
                                                     float* __restrict__ out) {
    __shared__ float red[256];
    const int t = blockIdx.x, tid = threadIdx.x;
    const float* xr = x + (size_t)t * Hdim;
    const float* ar = accbuf + (size_t)t * Hdim;
    float v0 = xr[tid] + ar[tid];
    float v1 = xr[tid + 256] + ar[tid + 256];
    float v2 = xr[tid + 512] + ar[tid + 512];
    red[tid] = v0 + v1 + v2;
    __syncthreads();
    for (int o = 128; o > 0; o >>= 1) { if (tid < o) red[tid] += red[tid + o]; __syncthreads(); }
    float mean = red[0] * (1.0f / Hdim);
    __syncthreads();
    float d0 = v0 - mean, d1 = v1 - mean, d2 = v2 - mean;
    red[tid] = d0 * d0 + d1 * d1 + d2 * d2;
    __syncthreads();
    for (int o = 128; o > 0; o >>= 1) { if (tid < o) red[tid] += red[tid + o]; __syncthreads(); }
    float rstd = rsqrtf(red[0] * (1.0f / Hdim) + 1e-5f);
    out[(size_t)t * Hdim + tid]       = d0 * rstd * gamma[tid]       + beta[tid];
    out[(size_t)t * Hdim + tid + 256] = d1 * rstd * gamma[tid + 256] + beta[tid + 256];
    out[(size_t)t * Hdim + tid + 512] = d2 * rstd * gamma[tid + 512] + beta[tid + 512];
}

extern "C" void kernel_launch(void* const* d_in, const int* in_sizes, int n_in,
                              void* d_out, int out_size, void* d_ws, size_t ws_size,
                              hipStream_t stream) {
    const float* x      = (const float*)d_in[0];
    const float* w1     = (const float*)d_in[1];
    const float* b1     = (const float*)d_in[2];
    const float* w2     = (const float*)d_in[3];
    const float* b2     = (const float*)d_in[4];
    const float* down_w = (const float*)d_in[5];
    const float* down_b = (const float*)d_in[6];
    const float* up_w   = (const float*)d_in[7];
    const float* gamma  = (const float*)d_in[8];
    const float* beta   = (const float*)d_in[9];
    float* out = (float*)d_out;

    // workspace carve-up
    char* ws = (char*)d_ws;
    size_t off = 0;
    auto take = [&](size_t bytes) { void* p = ws + off; off = (off + bytes + 255) & ~(size_t)255; return p; };
    bf16* xb        = (bf16*)take((size_t)BS * Hdim * 2);
    bf16* w1T       = (bf16*)take((size_t)H2 * Hdim * 2);
    float* h1       = (float*)take((size_t)BS * H2 * 4);
    bf16* dwT       = (bf16*)take((size_t)Edim * Hdim * Ddim * 2);
    bf16* upT       = (bf16*)take((size_t)Edim * Hdim * Ddim * 2);
    float* accbuf   = (float*)take((size_t)BS * Hdim * 4);
    int* counts     = (int*)take(Edim * 4);
    int* tok_list   = (int*)take((size_t)Edim * BS * 4);
    float* gate_list= (float*)take((size_t)Edim * BS * 4);

    hipMemsetAsync(counts, 0, Edim * sizeof(int), stream);
    hipMemsetAsync(accbuf, 0, (size_t)BS * Hdim * sizeof(float), stream);

    // conversions / transposes to bf16 N-major
    {
        int n = BS * Hdim;
        k_cvt_bf16<<<(n + 255) / 256, 256, 0, stream>>>(x, xb, n);
        k_cvt_transpose<<<(Hdim * H2 + 255) / 256, 256, 0, stream>>>(w1, w1T, Hdim, H2);
        dim3 g((Hdim * Ddim + 255) / 256, Edim);
        k_cvt_transpose_b<<<g, 256, 0, stream>>>(down_w, dwT, Hdim, Ddim); // [H][D]->[D][H]
        k_cvt_transpose_b<<<g, 256, 0, stream>>>(up_w, upT, Ddim, Hdim);   // [D][H]->[H][D]
    }

    // router GEMM (WMMA): 512 M-tiles x 3 (8 waves x 16-col tiles => 128 cols each)
    k_router_gemm<<<dim3(BS / 16, H2 / 128), 256, 0, stream>>>(xb, w1T, b1, h1);

    // logits + softmax + top2 + compaction
    k_router_topk<<<(BS + 255) / 256, 256, 0, stream>>>(h1, w2, b2, counts, tok_list, gate_list);

    // expert FFN (WMMA), grid covers worst-case token count per expert
    k_expert_ffn<<<dim3(BS / 16, Edim), 256, 0, stream>>>(xb, dwT, upT, down_b,
                                                          counts, tok_list, gate_list, accbuf);

    // residual + LayerNorm
    k_residual_ln<<<BS, 256, 0, stream>>>(x, accbuf, gamma, beta, out);
}
